// ChronosCore_10033043603917
// MI455X (gfx1250) — compile-verified, matrix operand order checked by reading, not run on hardware
//
#include <hip/hip_runtime.h>
#include <hip/hip_bf16.h>
#include <math.h>

// ---- model config (must match reference) ----
#define BB 2
#define TT 1024
#define VV 32000
#define DD 512
#define PP 128
#define SSLOT 16384
#define KDIM 64
#define VDIM 64
#define TOPK 4
#define HHID 512
#define LHID 512
#define HSTEPS 2
#define LSTEPS 2
#define MACDIM (DD + PP + TOPK * VDIM)   // 896
#define NROW (BB * TT)                   // 2048

typedef __attribute__((ext_vector_type(16))) __bf16 v16bf;
typedef __attribute__((ext_vector_type(8)))  float  v8f;

__device__ __forceinline__ unsigned short f2bf(float f) {
  union { float f; unsigned u; } v; v.f = f;
  unsigned r = v.u + 0x7FFFu + ((v.u >> 16) & 1u);   // round-to-nearest-even
  return (unsigned short)(r >> 16);
}

union FragU { v16bf v; uint4 q[2]; };

// WMMA 16x32 bf16 A-fragment from a row-major bf16 [*, ld] matrix.
// ISA 7.12.2: lane m = lane&15, khalf = lane>>4;
// elems 0..7 -> k = 8*khalf + 0..7 ; elems 8..15 -> k = 16 + 8*khalf + 0..7.
// B-fragments use the same mapping on an N-major (B^T) matrix (lane -> column n).
__device__ __forceinline__ v16bf load_frag(const unsigned short* base, int ld, int lane) {
  int m  = lane & 15;
  int kh = (lane >> 4) * 8;
  FragU f;
  const unsigned short* p = base + (size_t)m * ld + kh;
  f.q[0] = *(const uint4*)(p);
  f.q[1] = *(const uint4*)(p + 16);
  return f.v;
}

__device__ __forceinline__ v8f wmma_bf16(v16bf a, v16bf b, v8f c) {
  return __builtin_amdgcn_wmma_f32_16x16x32_bf16(false, a, false, b, (short)0, c, false, false);
}

// top-4 insert, sorted (val desc, idx asc) -> matches jax.lax.top_k tie-break
__device__ __forceinline__ void topk_insert(float v, int i, float bv[4], int bi[4]) {
  if (v < bv[3] || (v == bv[3] && i > bi[3])) return;
  bv[3] = v; bi[3] = i;
#pragma unroll
  for (int j = 3; j > 0; --j) {
    bool sw = (bv[j] > bv[j-1]) || (bv[j] == bv[j-1] && bi[j] < bi[j-1]);
    if (sw) {
      float tv = bv[j]; bv[j] = bv[j-1]; bv[j-1] = tv;
      int   ti = bi[j]; bi[j] = bi[j-1]; bi[j-1] = ti;
    }
  }
}

// ---------------- kernels ----------------

// X[b,t,:] = tok_emb[ids[b,t]] + pos_emb[t]  (stored bf16)
__global__ void k_embed(const int* __restrict__ ids, const float* __restrict__ tok,
                        const float* __restrict__ pos, unsigned short* __restrict__ XB) {
  int row = blockIdx.x;            // b*TT + t
  int t   = row & (TT - 1);
  int id  = ids[row];
  for (int j = threadIdx.x; j < DD; j += blockDim.x)
    XB[(size_t)row * DD + j] = f2bf(tok[(size_t)id * DD + j] + pos[(size_t)t * DD + j]);
}

__global__ void k_f2bf(const float* __restrict__ src, unsigned short* __restrict__ dst, int n) {
  int i = blockIdx.x * blockDim.x + threadIdx.x;
  if (i < n) dst[i] = f2bf(src[i]);
}

// src [rows,cols] f32 -> dst [cols,rows] bf16
__global__ void k_transpose_f2bf(const float* __restrict__ src, unsigned short* __restrict__ dst,
                                 int rows, int cols) {
  int i = blockIdx.x * blockDim.x + threadIdx.x;
  if (i < rows * cols) {
    int r = i / cols, c = i % cols;
    dst[(size_t)c * rows + r] = f2bf(src[i]);
  }
}

// C = A[M,K] @ Bt[N,K]^T ; one wave per 16x16 tile; K must be a multiple of 64.
// Two independent accumulation chains so both WMMAs per step are outstanding together.
// MODE 0: store bf16 ; MODE 1: store f32 with bias + exact gelu
template <int MODE>
__global__ void k_gemm_small(const unsigned short* __restrict__ A,
                             const unsigned short* __restrict__ Bt,
                             void* __restrict__ Cout, const float* __restrict__ bias,
                             int N, int K) {
  int lane = threadIdx.x;
  int m0 = blockIdx.x * 16, n0 = blockIdx.y * 16;
  const unsigned short* Ab = A  + (size_t)m0 * K;
  const unsigned short* Bb = Bt + (size_t)n0 * K;
  v8f c0 = {}, c1 = {};
  for (int k0 = 0; k0 < K; k0 += 64) {
    v16bf a0 = load_frag(Ab + k0,      K, lane);
    v16bf b0 = load_frag(Bb + k0,      K, lane);
    v16bf a1 = load_frag(Ab + k0 + 32, K, lane);
    v16bf b1 = load_frag(Bb + k0 + 32, K, lane);
    c0 = wmma_bf16(a0, b0, c0);
    c1 = wmma_bf16(a1, b1, c1);
  }
  int n  = n0 + (lane & 15);
  int mb = (lane >> 4) << 3;
#pragma unroll
  for (int r = 0; r < 8; ++r) {
    int m = m0 + mb + r;
    float v = c0[r] + c1[r];
    if (MODE == 1) {
      v += bias[n];
      v = 0.5f * v * (1.0f + erff(v * 0.70710678118654752f));  // exact gelu
      ((float*)Cout)[(size_t)m * N + n] = v;
    } else {
      ((unsigned short*)Cout)[(size_t)m * N + n] = f2bf(v);
    }
  }
}

// Fused sim = QB @ KeysB^T (WMMA) + per-row running top-4 + ltm_vals gather.
// One block = 16 rows, 8 waves; loops S in 128-column chunks.
#define SIMLD 132   // padded stride: 132 dwords -> rows offset 4 banks apart
__global__ void __launch_bounds__(256)
k_sim_topk(const unsigned short* __restrict__ QB,
           const unsigned short* __restrict__ KeysB,
           const float* __restrict__ ltm_vals,
           float* __restrict__ tv_out, int* __restrict__ idx_out) {
  __shared__ float simbuf[16 * SIMLD];
  __shared__ float candv[16 * 64];
  __shared__ int   candi[16 * 64];
  __shared__ int   topidx[16 * TOPK];
  int lane = threadIdx.x & 31, wave = threadIdx.x >> 5;
  int m0 = blockIdx.x * 16;

  // A fragments (K = 64 -> two k-steps), identical for every wave & chunk
  v16bf a0 = load_frag(QB + (size_t)m0 * KDIM + 0,  KDIM, lane);
  v16bf a1 = load_frag(QB + (size_t)m0 * KDIM + 32, KDIM, lane);

  float bv[4] = {-INFINITY, -INFINITY, -INFINITY, -INFINITY};
  int   bi[4] = {0x7fffffff, 0x7fffffff, 0x7fffffff, 0x7fffffff};
  int row = threadIdx.x >> 4;   // 0..15
  int sub = threadIdx.x & 15;   // 16 threads per row

  for (int ch = 0; ch < SSLOT / 128; ++ch) {
    int n0 = ch * 128 + wave * 16;
    v16bf b0 = load_frag(KeysB + (size_t)n0 * KDIM + 0,  KDIM, lane);
    v16bf b1 = load_frag(KeysB + (size_t)n0 * KDIM + 32, KDIM, lane);
    v8f c0 = {}, c1 = {};
    c0 = wmma_bf16(a0, b0, c0);              // independent chains
    c1 = wmma_bf16(a1, b1, c1);
    int n = lane & 15, mb = (lane >> 4) << 3;
#pragma unroll
    for (int r = 0; r < 8; ++r)
      simbuf[(mb + r) * SIMLD + wave * 16 + n] = c0[r] + c1[r];
    __syncthreads();
    // max-of-8 pre-filter: scan order is increasing index, so a value equal to
    // the current 4th-best always loses the tie-break -> skip unless strictly greater.
    float v0 = simbuf[row * SIMLD + sub * 8 + 0];
    float v1 = simbuf[row * SIMLD + sub * 8 + 1];
    float v2 = simbuf[row * SIMLD + sub * 8 + 2];
    float v3 = simbuf[row * SIMLD + sub * 8 + 3];
    float v4 = simbuf[row * SIMLD + sub * 8 + 4];
    float v5 = simbuf[row * SIMLD + sub * 8 + 5];
    float v6 = simbuf[row * SIMLD + sub * 8 + 6];
    float v7 = simbuf[row * SIMLD + sub * 8 + 7];
    float m8 = fmaxf(fmaxf(fmaxf(v0, v1), fmaxf(v2, v3)),
                     fmaxf(fmaxf(v4, v5), fmaxf(v6, v7)));
    if (m8 > bv[3]) {
      int base = ch * 128 + sub * 8;
      topk_insert(v0, base + 0, bv, bi); topk_insert(v1, base + 1, bv, bi);
      topk_insert(v2, base + 2, bv, bi); topk_insert(v3, base + 3, bv, bi);
      topk_insert(v4, base + 4, bv, bi); topk_insert(v5, base + 5, bv, bi);
      topk_insert(v6, base + 6, bv, bi); topk_insert(v7, base + 7, bv, bi);
    }
    __syncthreads();
  }

#pragma unroll
  for (int q = 0; q < 4; ++q) {
    candv[row * 64 + sub * 4 + q] = bv[q];
    candi[row * 64 + sub * 4 + q] = bi[q];
  }
  __syncthreads();
  if (threadIdx.x < 16) {
    int rr = threadIdx.x;
    float fv[4] = {-INFINITY, -INFINITY, -INFINITY, -INFINITY};
    int   fi[4] = {0x7fffffff, 0x7fffffff, 0x7fffffff, 0x7fffffff};
    for (int cnd = 0; cnd < 64; ++cnd)
      topk_insert(candv[rr * 64 + cnd], candi[rr * 64 + cnd], fv, fi);
#pragma unroll
    for (int q = 0; q < TOPK; ++q) {
      idx_out[(size_t)(m0 + rr) * TOPK + q] = fi[q];
      topidx[rr * TOPK + q] = fi[q];
    }
  }
  __syncthreads();
  // gather ltm_vals rows: tv_out[row, k, :VDIM]
  for (int e = threadIdx.x; e < 16 * TOPK * VDIM; e += blockDim.x) {
    int r = e >> 8, off = e & 255, q = off >> 6, dd = off & 63;
    tv_out[(size_t)(m0 + r) * (TOPK * VDIM) + off] =
        ltm_vals[(size_t)topidx[r * TOPK + q] * VDIM + dd];
  }
}

// mac = concat(tok(512), persistent(128), tv(256)) in bf16
__global__ void k_build_mac(const unsigned short* __restrict__ XB,
                            const float* __restrict__ persistent,
                            const float* __restrict__ tv,
                            unsigned short* __restrict__ MACB) {
  int row = blockIdx.x;
  for (int j = threadIdx.x; j < MACDIM; j += blockDim.x) {
    unsigned short v;
    if (j < DD)            v = XB[(size_t)row * DD + j];
    else if (j < DD + PP)  v = f2bf(persistent[j - DD]);
    else                   v = f2bf(tv[(size_t)row * (TOPK * VDIM) + (j - DD - PP)]);
    MACB[(size_t)row * MACDIM + j] = v;
  }
}

struct RecParams {
  const float *h_ir_w, *h_ir_b, *h_hr_w, *h_hr_b, *h_iz_w, *h_iz_b,
              *h_hz_w, *h_hz_b, *h_in_w, *h_in_b, *h_hn_w, *h_hn_b;
  const float *h2c_w, *h2c_b;
  const float *l_ir_w, *l_ir_b, *l_hr_w, *l_hr_b, *l_iz_w, *l_iz_b,
              *l_hz_w, *l_hz_b, *l_in_w, *l_in_b, *l_hn_w, *l_hn_b;
  const float *l2o_w, *l2o_b;
};

// Sequential recurrence: one workgroup per batch chain, 512 threads,
// thread j owns output column j of every GEMV (coalesced weight reads; all
// recurrent weights (~20MB) stay resident in the 192MB L2). f32 for state
// stability; M=2 rows makes WMMA tiles 8x wasted here, VALU is the right path.
__global__ void __launch_bounds__(512)
k_recurrent(float* __restrict__ ENC, RecParams p) {
  int b = blockIdx.x, j = threadIdx.x;
  __shared__ float xe[DD], hh[HHID], ll[LHID], cx[DD];
  hh[j] = 0.f; ll[j] = 0.f;
  __syncthreads();
  for (int t = 0; t < TT; ++t) {
    float* erow = ENC + ((size_t)b * TT + t) * DD;
    xe[j] = erow[j];
    __syncthreads();
    for (int hs = 0; hs < HSTEPS; ++hs) {
      // ---- GRU h : thread j computes all 6 dot products for column j
      float ar = p.h_ir_b[j], az = p.h_iz_b[j], an = p.h_in_b[j];
      float hr = p.h_hr_b[j], hz = p.h_hz_b[j], hn = p.h_hn_b[j];
      for (int k = 0; k < DD; ++k) {
        float xv = xe[k], hv = hh[k];
        ar += xv * p.h_ir_w[k * HHID + j];
        az += xv * p.h_iz_w[k * HHID + j];
        an += xv * p.h_in_w[k * HHID + j];
        hr += hv * p.h_hr_w[k * HHID + j];
        hz += hv * p.h_hz_w[k * HHID + j];
        hn += hv * p.h_hn_w[k * HHID + j];
      }
      float r = 1.f / (1.f + expf(-(ar + hr)));
      float z = 1.f / (1.f + expf(-(az + hz)));
      float n = tanhf(an + r * hn);
      float hnew = (1.f - z) * n + z * hh[j];
      __syncthreads();
      hh[j] = hnew;
      __syncthreads();
      // ---- ctx = h @ h2c + b
      float c = p.h2c_b[j];
      for (int k = 0; k < HHID; ++k) c += hh[k] * p.h2c_w[k * DD + j];
      cx[j] = c;
      __syncthreads();
      // ---- GRU l (input = concat(xe, cx))
      for (int ls = 0; ls < LSTEPS; ++ls) {
        float lar = p.l_ir_b[j], laz = p.l_iz_b[j], lan = p.l_in_b[j];
        float lhr = p.l_hr_b[j], lhz = p.l_hz_b[j], lhn = p.l_hn_b[j];
        for (int k = 0; k < DD; ++k) {
          float xv = xe[k];
          lar += xv * p.l_ir_w[k * LHID + j];
          laz += xv * p.l_iz_w[k * LHID + j];
          lan += xv * p.l_in_w[k * LHID + j];
        }
        for (int k = 0; k < DD; ++k) {
          float cv = cx[k];
          lar += cv * p.l_ir_w[(DD + k) * LHID + j];
          laz += cv * p.l_iz_w[(DD + k) * LHID + j];
          lan += cv * p.l_in_w[(DD + k) * LHID + j];
        }
        for (int k = 0; k < LHID; ++k) {
          float lv = ll[k];
          lhr += lv * p.l_hr_w[k * LHID + j];
          lhz += lv * p.l_hz_w[k * LHID + j];
          lhn += lv * p.l_hn_w[k * LHID + j];
        }
        float lr  = 1.f / (1.f + expf(-(lar + lhr)));
        float lz  = 1.f / (1.f + expf(-(laz + lhz)));
        float lnv = tanhf(lan + lr * lhn);
        float lnew = (1.f - lz) * lnv + lz * ll[j];
        __syncthreads();
        ll[j] = lnew;
        __syncthreads();
      }
      // ---- enc = enc + l @ l2o + b
      float eo = p.l2o_b[j];
      for (int k = 0; k < LHID; ++k) eo += ll[k] * p.l2o_w[k * DD + j];
      float xnew = xe[j] + eo;
      __syncthreads();
      xe[j] = xnew;
      __syncthreads();
    }
    erow[j] = xe[j];
    __syncthreads();
  }
}

// LayerNorm over D, emit bf16 for the logits GEMM
__global__ void k_layernorm(const float* __restrict__ ENC, const float* __restrict__ g,
                            const float* __restrict__ beta, unsigned short* __restrict__ FEB) {
  __shared__ float red[256];
  int row = blockIdx.x, tid = threadIdx.x;
  const float* x = ENC + (size_t)row * DD;
  float x0 = x[tid], x1 = x[tid + 256];
  red[tid] = x0 + x1;
  __syncthreads();
  for (int o = 128; o > 0; o >>= 1) { if (tid < o) red[tid] += red[tid + o]; __syncthreads(); }
  float mu = red[0] * (1.0f / DD);
  __syncthreads();
  float d0 = x0 - mu, d1 = x1 - mu;
  red[tid] = d0 * d0 + d1 * d1;
  __syncthreads();
  for (int o = 128; o > 0; o >>= 1) { if (tid < o) red[tid] += red[tid + o]; __syncthreads(); }
  float rstd = rsqrtf(red[0] * (1.0f / DD) + 1e-5f);
  FEB[(size_t)row * DD + tid]       = f2bf(d0 * rstd * g[tid] + beta[tid]);
  FEB[(size_t)row * DD + tid + 256] = f2bf(d1 * rstd * g[tid + 256] + beta[tid + 256]);
}

// logits = FEB[2048,512] @ TokB[32000,512]^T
// 4 waves/block; each wave computes a 16x64 strip (4 accumulators): one A fragment
// feeds 4 independent WMMAs per k-step, 8 b128 B-loads in flight per wait.
// A tile staged in LDS with padded stride 520 (1040B = 4 dwords mod 64 banks)
// so the 16 lanes of a fragment load hit distinct bank quads.
#define ALD 520
__global__ void __launch_bounds__(128)
k_gemm_logits(const unsigned short* __restrict__ FEB, const unsigned short* __restrict__ TokB,
              float* __restrict__ out) {
  __shared__ __align__(16) unsigned short As[16 * ALD];
  int lane = threadIdx.x & 31, wave = threadIdx.x >> 5;
  int m0 = blockIdx.x * 16;
  int nb = blockIdx.y * 256 + wave * 64;     // this wave's 4 column tiles
  // stage A tile (16x512) into padded LDS
  const uint4* src = (const uint4*)(FEB + (size_t)m0 * DD);
  for (int i = threadIdx.x; i < 16 * DD / 8; i += 128) {
    int r = i >> 6, cidx = i & 63;           // 64 uint4 per source row
    ((uint4*)As)[r * (ALD / 8) + cidx] = src[i];
  }
  __syncthreads();
  const unsigned short* B0 = TokB + (size_t)(nb +  0) * DD;
  const unsigned short* B1 = TokB + (size_t)(nb + 16) * DD;
  const unsigned short* B2 = TokB + (size_t)(nb + 32) * DD;
  const unsigned short* B3 = TokB + (size_t)(nb + 48) * DD;
  v8f c0 = {}, c1 = {}, c2 = {}, c3 = {};
  for (int k0 = 0; k0 < DD; k0 += 32) {
    if (k0 + 32 < DD)
      __builtin_prefetch(B0 + k0 + 32, 0, 0);          // global_prefetch_b8
    v16bf a  = load_frag(As + k0, ALD, lane);
    v16bf b0 = load_frag(B0 + k0, DD, lane);
    v16bf b1 = load_frag(B1 + k0, DD, lane);
    v16bf b2 = load_frag(B2 + k0, DD, lane);
    v16bf b3 = load_frag(B3 + k0, DD, lane);
    c0 = wmma_bf16(a, b0, c0);
    c1 = wmma_bf16(a, b1, c1);
    c2 = wmma_bf16(a, b2, c2);
    c3 = wmma_bf16(a, b3, c3);
  }
  int n  = lane & 15;
  int mb = (lane >> 4) << 3;
#pragma unroll
  for (int r = 0; r < 8; ++r) {
    size_t rowoff = (size_t)(m0 + mb + r) * VV + nb + n;
    out[rowoff +  0] = c0[r];
    out[rowoff + 16] = c1[r];
    out[rowoff + 32] = c2[r];
    out[rowoff + 48] = c3[r];
  }
}

// ---------------- launch ----------------

extern "C" void kernel_launch(void* const* d_in, const int* in_sizes, int n_in,
                              void* d_out, int out_size, void* d_ws, size_t ws_size,
                              hipStream_t stream) {
  (void)in_sizes; (void)n_in; (void)out_size; (void)ws_size;
  const int*   ids        = (const int*)d_in[0];
  const float* tok_emb    = (const float*)d_in[1];
  const float* pos_emb    = (const float*)d_in[2];
  const float* persistent = (const float*)d_in[3];
  const float* ltm_keys   = (const float*)d_in[4];
  const float* ltm_vals   = (const float*)d_in[5];
  const float* qproj_w    = (const float*)d_in[6];
  const float* in_proj_w  = (const float*)d_in[7];
  const float* in_proj_b  = (const float*)d_in[8];
  RecParams rp;
  rp.h_ir_w = (const float*)d_in[9];  rp.h_ir_b = (const float*)d_in[10];
  rp.h_hr_w = (const float*)d_in[11]; rp.h_hr_b = (const float*)d_in[12];
  rp.h_iz_w = (const float*)d_in[13]; rp.h_iz_b = (const float*)d_in[14];
  rp.h_hz_w = (const float*)d_in[15]; rp.h_hz_b = (const float*)d_in[16];
  rp.h_in_w = (const float*)d_in[17]; rp.h_in_b = (const float*)d_in[18];
  rp.h_hn_w = (const float*)d_in[19]; rp.h_hn_b = (const float*)d_in[20];
  rp.h2c_w  = (const float*)d_in[21]; rp.h2c_b  = (const float*)d_in[22];
  rp.l_ir_w = (const float*)d_in[23]; rp.l_ir_b = (const float*)d_in[24];
  rp.l_hr_w = (const float*)d_in[25]; rp.l_hr_b = (const float*)d_in[26];
  rp.l_iz_w = (const float*)d_in[27]; rp.l_iz_b = (const float*)d_in[28];
  rp.l_hz_w = (const float*)d_in[29]; rp.l_hz_b = (const float*)d_in[30];
  rp.l_in_w = (const float*)d_in[31]; rp.l_in_b = (const float*)d_in[32];
  rp.l_hn_w = (const float*)d_in[33]; rp.l_hn_b = (const float*)d_in[34];
  rp.l2o_w  = (const float*)d_in[35]; rp.l2o_b  = (const float*)d_in[36];
  const float* ln_g = (const float*)d_in[37];
  const float* ln_b = (const float*)d_in[38];

  // outputs: logits f32 [B,T,V] | tvs f32 [B,T,K,VD] | idxs i32 [B,T,K]
  float* out_logits = (float*)d_out;
  float* out_tv     = out_logits + (size_t)NROW * VV;
  int*   out_idx    = (int*)(out_tv + (size_t)NROW * TOPK * VDIM);

  // workspace carve-out (~48MB)
  char* w = (char*)d_ws;
  auto alloc = [&](size_t bytes) -> void* {
    void* pp = w; w += (bytes + 255) & ~(size_t)255; return pp;
  };
  unsigned short* XB     = (unsigned short*)alloc((size_t)NROW * DD * 2);
  unsigned short* KeysB  = (unsigned short*)alloc((size_t)SSLOT * KDIM * 2);
  unsigned short* TokB   = (unsigned short*)alloc((size_t)VV * DD * 2);
  unsigned short* QprojT = (unsigned short*)alloc((size_t)KDIM * DD * 2);
  unsigned short* WinT   = (unsigned short*)alloc((size_t)DD * MACDIM * 2);
  unsigned short* QB     = (unsigned short*)alloc((size_t)NROW * KDIM * 2);
  unsigned short* MACB   = (unsigned short*)alloc((size_t)NROW * MACDIM * 2);
  float*          ENC    = (float*)alloc((size_t)NROW * DD * 4);
  unsigned short* FEB    = (unsigned short*)alloc((size_t)NROW * DD * 2);

  // 1) embeddings -> bf16
  k_embed<<<NROW, 256, 0, stream>>>(ids, tok_emb, pos_emb, XB);
  // 2) bf16 weight copies / transposes
  k_f2bf<<<(SSLOT * KDIM + 255) / 256, 256, 0, stream>>>(ltm_keys, KeysB, SSLOT * KDIM);
  k_f2bf<<<(VV * DD + 255) / 256, 256, 0, stream>>>(tok_emb, TokB, VV * DD);
  k_transpose_f2bf<<<(DD * KDIM + 255) / 256, 256, 0, stream>>>(qproj_w, QprojT, DD, KDIM);
  k_transpose_f2bf<<<(MACDIM * DD + 255) / 256, 256, 0, stream>>>(in_proj_w, WinT, MACDIM, DD);
  // 3) q = X @ qproj  (WMMA, -> bf16)
  k_gemm_small<0><<<dim3(NROW / 16, KDIM / 16), 32, 0, stream>>>(XB, QprojT, QB, nullptr, KDIM, DD);
  // 4) fused sim (WMMA) + top-k + gather
  k_sim_topk<<<NROW / 16, 256, 0, stream>>>(QB, KeysB, ltm_vals, out_tv, out_idx);
  // 5) mac concat (bf16)
  k_build_mac<<<NROW, 256, 0, stream>>>(XB, persistent, out_tv, MACB);
  // 6) enc0 = gelu(mac @ in_proj + b)  (WMMA, f32 out)
  k_gemm_small<1><<<dim3(NROW / 16, DD / 16), 32, 0, stream>>>(MACB, WinT, ENC, in_proj_b, DD, MACDIM);
  // 7) sequential recurrence, in-place on ENC
  k_recurrent<<<BB, 512, 0, stream>>>(ENC, rp);
  // 8) layernorm -> bf16
  k_layernorm<<<NROW, 256, 0, stream>>>(ENC, ln_g, ln_b, FEB);
  // 9) logits = fe @ tok_emb^T  (WMMA, dominant 67 GFLOP GEMM; 16x64 per wave)
  k_gemm_logits<<<dim3(NROW / 16, VV / 256), 128, 0, stream>>>(FEB, TokB, out_logits);
}
